// AudioMamba1Model_36524401885834
// MI455X (gfx1250) — compile-verified, matrix operand order checked by reading, not run on hardware
//
#include <hip/hip_runtime.h>

typedef _Float16 half_t;
typedef half_t v16h __attribute__((ext_vector_type(16)));
typedef half_t h2 __attribute__((ext_vector_type(2)));
typedef float  v8f  __attribute__((ext_vector_type(8)));

#define IN_DIM 36
#define HID 8
#define AUD 32
#define DIN 24        // D_INNER
#define DST 4         // D_STATE

#define THREADS 256
#define WAVES 8
#define TROWS 16
#define BROWS 128

// f16 staging buffers per wave, each 512 halves (exact 16x16x32 A-operand layout)
#define SB_XA0 0
#define SB_XA1 1
#define SB_U   2
#define SB_XI  3
#define SB_Y   4

// f32 pool per wave
#define FPW 576
#define SZ_OFF 0      // silu(z) 16x24
#define XDB_OFF 384   // xdb 16x12 (cols 0..8 + BC at col 9)

// const table offsets (floats)
#define C_CW 0
#define C_CB 24
#define C_DTW 48
#define C_DTB 72
#define C_DP 96
#define C_FINB 120
#define C_FOUTB 128

// K index of element j in a 16x32 f16 A/B operand for half-wave group g
__device__ __forceinline__ int wmma_kk(int g, int j) {
  const int v = j >> 1, h = j & 1;
  return (v < 4) ? (8 * g + 2 * v + h) : (16 + 8 * g + 2 * (v - 4) + h);
}
// inverse: half-index inside a 512-half A-layout buffer for element (row m, k)
__device__ __forceinline__ int alay(int m, int k) {
  const int g = (k >> 3) & 1;
  const int j = (k & 7) + 8 * ((k >> 4) & 1);
  return (m + 16 * g) * 16 + j;
}

// wave-private LDS produce->consume sync (lockstep wave32: drain DScnt only)
__device__ __forceinline__ void wave_lds_sync() {
  asm volatile("s_wait_dscnt 0x0" ::: "memory");
}

// one-time B operand from global row-major (N,K) weight, branch-free loads
__device__ __forceinline__ v16h buildB_g(const float* W, int K, int n, int g,
                                         int kmax, int nmax) {
  v16h b;
#pragma unroll
  for (int j = 0; j < 16; ++j) {
    const int k = wmma_kk(g, j);
    const bool valid = (n < nmax) && (k < kmax);
    const float w = W[valid ? n * K + k : 0];
    b[j] = valid ? (half_t)w : (half_t)0.f;
  }
  return b;
}
// B operand from zero-padded LDS table with row stride 32 (fully unconditional)
__device__ __forceinline__ v16h buildB_l(const float* Wp, int n, int g) {
  v16h b;
#pragma unroll
  for (int j = 0; j < 16; ++j) b[j] = (half_t)Wp[n * 32 + wmma_kk(g, j)];
  return b;
}

__device__ __forceinline__ float siluf(float v) {
  return v * __builtin_amdgcn_rcpf(1.f + __expf(-v));
}
__device__ __forceinline__ float softplusf(float v) {
  return (v > 20.f) ? v : __logf(1.f + __expf(v));
}

__global__ __launch_bounds__(THREADS)
void mamba_fused(const float* __restrict__ x,
                 const float* __restrict__ f_in_w, const float* __restrict__ f_in_b,
                 const float* __restrict__ f_out_w, const float* __restrict__ f_out_b,
                 const float* __restrict__ in_proj_w,
                 const float* __restrict__ conv_w, const float* __restrict__ conv_b,
                 const float* __restrict__ x_proj_w,
                 const float* __restrict__ dt_proj_w, const float* __restrict__ dt_proj_b,
                 const float* __restrict__ Dp, const float* __restrict__ out_proj_w,
                 float* __restrict__ out, int nrows, int ntiles) {
  __shared__ __align__(32) half_t stg[WAVES][5][512];  // 40960 B
  __shared__ __align__(16) float fpool[WAVES][FPW];    // 18432 B
  __shared__ float cbufp[AUD * 32];                    //  4096 B (stride-32 padded)
  __shared__ __align__(16) float scon[160];            //   640 B

  const int tid  = threadIdx.x;
  const int wave = tid >> 5;
  const int lane = tid & 31;
  const int g = lane >> 4;
  const int r = lane & 15;
  float* FP = fpool[wave];

  // ---- one-time init: zero f16 staging (pads persist as zero) ----
  {
    unsigned* z = (unsigned*)&stg[0][0][0];
    for (int i = tid; i < WAVES * 5 * 512 / 2; i += THREADS) z[i] = 0u;
  }
  // zero pad columns of combined weight (cols 24..31)
  for (int i = tid; i < AUD * 8; i += THREADS) cbufp[(i >> 3) * 32 + 24 + (i & 7)] = 0.f;
  // combined[o][d] = sum_h f_out_w[o,h] * out_proj_w[h,d]
  for (int idx = tid; idx < AUD * DIN; idx += THREADS) {
    const int o = idx / DIN, d = idx % DIN;
    float s = 0.f;
#pragma unroll
    for (int h = 0; h < HID; ++h) s += f_out_w[o * HID + h] * out_proj_w[h * DIN + d];
    cbufp[o * 32 + d] = s;
  }
  // small per-channel constants
  if (tid < DIN) {
    scon[C_CW + tid]  = conv_w[2 * tid + 1];
    scon[C_CB + tid]  = conv_b[tid];
    scon[C_DTW + tid] = dt_proj_w[tid];
    scon[C_DTB + tid] = dt_proj_b[tid];
    scon[C_DP + tid]  = Dp[tid];
  }
  if (tid < HID) scon[C_FINB + tid] = f_in_b[tid];
  if (tid < AUD) scon[C_FOUTB + tid] = f_out_b[tid];
  __syncthreads();

  // ---- hoisted per-lane constants and weight B operands ----
  const float finb  = scon[C_FINB + (r < HID ? r : 0)];
  const float cw0 = scon[C_CW + r],     cb0 = scon[C_CB + r];
  const float cw1 = scon[C_CW + ((16 + r) % DIN)], cb1 = scon[C_CB + ((16 + r) % DIN)];
  const float bias0 = scon[C_FOUTB + r], bias1 = scon[C_FOUTB + 16 + r];

  const v16h bfin0 = buildB_g(f_in_w, IN_DIM, r, g, 32, HID);           // k 0..31
  v16h bfin1;                                                            // k 32..35
#pragma unroll
  for (int j = 0; j < 16; ++j) {
    const int k = 32 + wmma_kk(g, j);
    const bool valid = (r < HID) && (k < IN_DIM);
    const float w = f_in_w[valid ? r * IN_DIM + k : 0];
    bfin1[j] = valid ? (half_t)w : (half_t)0.f;
  }
  const v16h bip0 = buildB_g(in_proj_w, HID, r,      g, HID, 48);
  const v16h bip1 = buildB_g(in_proj_w, HID, 16 + r, g, HID, 48);
  const v16h bip2 = buildB_g(in_proj_w, HID, 32 + r, g, HID, 48);
  const v16h bxp  = buildB_g(x_proj_w, DIN, r, g, DIN, 9);
  const v16h bc0  = buildB_l(cbufp, r, g);
  const v16h bc1  = buildB_l(cbufp, 16 + r, g);

  for (int tile = blockIdx.x; tile < ntiles; tile += gridDim.x) {
    const int row0 = tile * BROWS;
    const bool full = (row0 + BROWS) <= nrows;
    __syncthreads();  // protect xa buffers across iterations (cross-wave writes)

    // coalesced float2 x load -> packed f16 pair -> single b32 LDS store
    // (clamped row index: fully unconditional loads)
#pragma unroll
    for (int it = 0; it < (BROWS * IN_DIM / 2) / THREADS; ++it) {  // 9
      const int pidx = tid + THREADS * it;
      const int rr = pidx / (IN_DIM / 2);
      const int kp = (pidx % (IN_DIM / 2)) * 2;
      int grow = row0 + rr;
      grow = grow < nrows ? grow : (nrows - 1);
      const float2 v = *(const float2*)&x[grow * IN_DIM + kp];
      const int wv = rr >> 4, m = rr & 15;
      const int chunk = kp >> 5;  // SB_XA0 / SB_XA1
      h2 p; p.x = (half_t)v.x; p.y = (half_t)v.y;
      *(h2*)&stg[wv][chunk][alay(m, kp & 31)] = p;
    }
    __syncthreads();

    // ---- stage 1: u = x @ f_in_w.T + f_in_b ----
    {
      const v16h a0 = *(const v16h*)&stg[wave][SB_XA0][lane * 16];
      const v16h a1 = *(const v16h*)&stg[wave][SB_XA1][lane * 16];
      v8f c = {};
      c = __builtin_amdgcn_wmma_f32_16x16x32_f16(false, a0, false, bfin0, (short)0, c, false, false);
      c = __builtin_amdgcn_wmma_f32_16x16x32_f16(false, a1, false, bfin1, (short)0, c, false, false);
      if (r < HID) {
#pragma unroll
        for (int i = 0; i < 8; ++i)
          stg[wave][SB_U][(i + 8 * g) * 16 + r] = (half_t)(c[i] + finb);
      }
    }
    wave_lds_sync();

    // ---- stage 2: xz = u @ in_proj_w.T (3 N-tiles) ----
    v8f x0, x1, x2;
    {
      const v16h a = *(const v16h*)&stg[wave][SB_U][lane * 16];
      v8f c0 = {}, c1 = {}, c2 = {};
      c0 = __builtin_amdgcn_wmma_f32_16x16x32_f16(false, a, false, bip0, (short)0, c0, false, false);
      c1 = __builtin_amdgcn_wmma_f32_16x16x32_f16(false, a, false, bip1, (short)0, c1, false, false);
      c2 = __builtin_amdgcn_wmma_f32_16x16x32_f16(false, a, false, bip2, (short)0, c2, false, false);
      x0 = c0; x1 = c1; x2 = c2;
    }
    // ---- stage 3: conv(L=1)+silu on xi; silu on z ----
#pragma unroll
    for (int i = 0; i < 8; ++i) {
      const int m = i + 8 * g;
      stg[wave][SB_XI][alay(m, r)] = (half_t)siluf(cw0 * x0[i] + cb0);   // d = r
      if (r < 8) {                                                       // d = 16+r
        stg[wave][SB_XI][alay(m, 16 + r)] = (half_t)siluf(cw1 * x1[i] + cb1);
      } else {                                                           // z ch r-8
        FP[SZ_OFF + m * DIN + (r - 8)] = siluf(x1[i]);
      }
      FP[SZ_OFF + m * DIN + 8 + r] = siluf(x2[i]);                       // z ch 8+r
    }
    wave_lds_sync();

    // ---- stage 4: xdb = xi_c @ x_proj_w.T (16x9) ----
    {
      const v16h a = *(const v16h*)&stg[wave][SB_XI][lane * 16];
      v8f c = {};
      c = __builtin_amdgcn_wmma_f32_16x16x32_f16(false, a, false, bxp, (short)0, c, false, false);
      if (r < 9) {
#pragma unroll
        for (int i = 0; i < 8; ++i) FP[XDB_OFF + (i + 8 * g) * 12 + r] = c[i];
      }
    }
    wave_lds_sync();

    // ---- stage 5: per-row BC = dot(B,C) ----
    if (lane < 16) {
      const float* rw = FP + XDB_OFF + lane * 12;
      float bc = 0.f;
#pragma unroll
      for (int n = 0; n < DST; ++n) bc += rw[1 + n] * rw[5 + n];
      FP[XDB_OFF + lane * 12 + 9] = bc;
    }
    wave_lds_sync();

    // ---- stage 6: y = xi_c*(softplus(dt*w+b)*BC + Dp) * silu(z) ----
    // lane owns row m=r and channel pairs d = 4*it + 2*g (packed h2 I/O)
    {
      const float dtraw = FP[XDB_OFF + r * 12 + 0];
      const float bc    = FP[XDB_OFF + r * 12 + 9];
#pragma unroll
      for (int it = 0; it < 6; ++it) {
        const int d = 4 * it + 2 * g;
        const float2 dtw = *(const float2*)&scon[C_DTW + d];
        const float2 dtb = *(const float2*)&scon[C_DTB + d];
        const float2 dpp = *(const float2*)&scon[C_DP + d];
        const float dts0 = softplusf(dtraw * dtw.x + dtb.x);
        const float dts1 = softplusf(dtraw * dtw.y + dtb.y);
        const h2 xp = *(const h2*)&stg[wave][SB_XI][alay(r, d)];
        const float2 szv = *(const float2*)&FP[SZ_OFF + r * DIN + d];
        const float y0 = (float)xp.x * (dts0 * bc + dpp.x) * szv.x;
        const float y1 = (float)xp.y * (dts1 * bc + dpp.y) * szv.y;
        h2 yp; yp.x = (half_t)y0; yp.y = (half_t)y1;
        *(h2*)&stg[wave][SB_Y][alay(r, d)] = yp;
      }
    }
    wave_lds_sync();

    // ---- stage 7: out = y @ combined.T + f_out_b; softmax via shuffles ----
    {
      const v16h a = *(const v16h*)&stg[wave][SB_Y][lane * 16];
      v8f c0 = {}, c1 = {};
      c0 = __builtin_amdgcn_wmma_f32_16x16x32_f16(false, a, false, bc0, (short)0, c0, false, false);
      c1 = __builtin_amdgcn_wmma_f32_16x16x32_f16(false, a, false, bc1, (short)0, c1, false, false);
      const int wrow0 = row0 + wave * TROWS;
#pragma unroll
      for (int i = 0; i < 8; ++i) {
        float a0 = c0[i] + bias0, a1 = c1[i] + bias1;
        float mx = fmaxf(a0, a1);
        mx = fmaxf(mx, __shfl_xor(mx, 1, 32));
        mx = fmaxf(mx, __shfl_xor(mx, 2, 32));
        mx = fmaxf(mx, __shfl_xor(mx, 4, 32));
        mx = fmaxf(mx, __shfl_xor(mx, 8, 32));
        const float e0 = __expf(a0 - mx), e1 = __expf(a1 - mx);
        float s = e0 + e1;
        s += __shfl_xor(s, 1, 32);
        s += __shfl_xor(s, 2, 32);
        s += __shfl_xor(s, 4, 32);
        s += __shfl_xor(s, 8, 32);
        const float inv = __builtin_amdgcn_rcpf(s);
        const int grow = wrow0 + i + 8 * g;
        if (full) {
          out[grow * AUD + r]      = e0 * inv;
          out[grow * AUD + 16 + r] = e1 * inv;
        } else if (grow < nrows) {
          out[grow * AUD + r]      = e0 * inv;
          out[grow * AUD + 16 + r] = e1 * inv;
        }
      }
    }
  }
}

extern "C" void kernel_launch(void* const* d_in, const int* in_sizes, int n_in,
                              void* d_out, int out_size, void* d_ws, size_t ws_size,
                              hipStream_t stream) {
  const float* x         = (const float*)d_in[0];
  const float* f_in_w    = (const float*)d_in[1];
  const float* f_in_b    = (const float*)d_in[2];
  const float* f_out_w   = (const float*)d_in[3];
  const float* f_out_b   = (const float*)d_in[4];
  const float* in_proj_w = (const float*)d_in[5];
  const float* conv_w    = (const float*)d_in[6];
  const float* conv_b    = (const float*)d_in[7];
  const float* x_proj_w  = (const float*)d_in[8];
  const float* dt_proj_w = (const float*)d_in[9];
  const float* dt_proj_b = (const float*)d_in[10];
  // d_in[11] = A_log: dead (L=1, h0=0)
  const float* Dp        = (const float*)d_in[12];
  const float* out_proj_w= (const float*)d_in[13];
  float* out = (float*)d_out;

  const int nrows  = in_sizes[0] / IN_DIM;
  const int ntiles = (nrows + BROWS - 1) / BROWS;
  const int blocks = ntiles < 1024 ? ntiles : 1024;

  hipLaunchKernelGGL(mamba_fused, dim3(blocks), dim3(THREADS), 0, stream,
                     x, f_in_w, f_in_b, f_out_w, f_out_b, in_proj_w,
                     conv_w, conv_b, x_proj_w, dt_proj_w, dt_proj_b,
                     Dp, out_proj_w, out, nrows, ntiles);
}
